// PaddedNaryEmitter_32899449487373
// MI455X (gfx1250) — matrix operand order repaired and checked by reading, not compiled, not run on hardware
//
#include <hip/hip_runtime.h>
#include <math.h>
#include <stdint.h>

// ---------------------------------------------------------------------------
// Types for CDNA5 WMMA (wave32, v_wmma_f32_16x16x32_bf16)
// ---------------------------------------------------------------------------
typedef __bf16 bf16;
typedef __bf16 v16bf __attribute__((ext_vector_type(16)));
typedef __bf16 v8bf  __attribute__((ext_vector_type(8)));
typedef __bf16 v4bf  __attribute__((ext_vector_type(4)));
typedef float  v8f   __attribute__((ext_vector_type(8)));
typedef int    v4i   __attribute__((ext_vector_type(4)));

__device__ __forceinline__ v16bf cat16(v8bf lo, v8bf hi) {
  return __builtin_shufflevector(lo, hi, 0,1,2,3,4,5,6,7,8,9,10,11,12,13,14,15);
}
__device__ __forceinline__ v8f wmma_bf16(v16bf a, v16bf b, v8f c) {
  return __builtin_amdgcn_wmma_f32_16x16x32_bf16(false, a, false, b, (short)0, c,
                                                 false, false);
}

// ---- gfx1250 async load-to-LDS (guarded: falls back to plain loads) -------
#define HAVE_ASYNC_LDS 0
#if defined(__has_builtin)
#if __has_builtin(__builtin_amdgcn_global_load_async_to_lds_b128) && \
    __has_builtin(__builtin_amdgcn_s_wait_asynccnt)
#undef HAVE_ASYNC_LDS
#define HAVE_ASYNC_LDS 1
#endif
#endif

#if HAVE_ASYNC_LDS
typedef __attribute__((address_space(1))) v4i gv4i;
typedef __attribute__((address_space(3))) v4i sv4i;
// Flat pointers to LDS carry the DS byte-offset in their low 32 bits
// global flat == AS1 numerically.
__device__ __forceinline__ sv4i* lds_cast(void* p) {
  return (sv4i*)(uint32_t)(uintptr_t)p;
}
__device__ __forceinline__ gv4i* gbl_cast(const void* p) {
  return (gv4i*)(uintptr_t)p;
}
#endif

#define H 1024
#define T_SEQ 2048
#define NB 64
#define CODEBOOK 2000

// ---------------------------------------------------------------------------
// K0: transpose + convert weight W[k][n] (f32) -> Wt[n][k] (bf16)
// grid (32,32), block (32,8)
// ---------------------------------------------------------------------------
__global__ void wt_convert_kernel(const float* __restrict__ W,
                                  bf16* __restrict__ Wt) {
  __shared__ float tile[32][33];
  const int bx = blockIdx.x, by = blockIdx.y;
  const int tx = threadIdx.x, ty = threadIdx.y;
  for (int i = 0; i < 4; ++i)
    tile[ty + i * 8][tx] = W[(size_t)(by * 32 + ty + i * 8) * H + bx * 32 + tx];
  __syncthreads();
  for (int i = 0; i < 4; ++i)
    Wt[(size_t)(bx * 32 + ty + i * 8) * H + by * 32 + tx] =
        (bf16)tile[tx][ty + i * 8];
}

// ---------------------------------------------------------------------------
// K1: fused dual GEMM  C0/C1 bf16[R x H] = X_f32[R x H] * Wt0/Wt1 (bf16 [n][k])
// One A-stage (fp32 -> bf16 LDS) feeds both weight matrices: halves X traffic
// and doubles WMMA work per barrier (16 v_wmma per K-step per wave).
// block = 256 threads (8 waves), tile 128(M) x 128(N) x2, K-step 32
// grid = (H/128, R/128)
// ---------------------------------------------------------------------------
#define AS_STR 40  // bf16 elems per LDS row (80B: 16B-aligned, conflict-free)

__global__ __launch_bounds__(256) void gemm_kv_xf32(
    const float* __restrict__ X, const bf16* __restrict__ Wt0,
    const bf16* __restrict__ Wt1, bf16* __restrict__ C0,
    bf16* __restrict__ C1) {
  __shared__ bf16 As[128 * AS_STR];
  __shared__ bf16 Bs[2][128 * AS_STR];
  const int tid = threadIdx.x;
  const int wave = tid >> 5, lane = tid & 31;
  const int waveM = wave & 3, waveN = wave >> 2;
  const int lane16 = lane & 15;
  const int hiHalf = lane >> 4;  // 0 or 1
  const int m0 = blockIdx.y * 128;
  const int n0 = blockIdx.x * 128;

  const int arow = tid >> 3;          // 0..31
  const int acol = (tid & 7) * 4;     // 0..28

  v8f acc[2][2][4];  // [weight][fm][fn]
  for (int w = 0; w < 2; ++w)
    for (int fm = 0; fm < 2; ++fm)
      for (int fn = 0; fn < 4; ++fn) acc[w][fm][fn] = (v8f)(0.0f);

  for (int k0 = 0; k0 < H; k0 += 32) {
    // --- stage A: 128x32 fp32 -> bf16 LDS ---
    for (int i = 0; i < 4; ++i) {
      const int r = arow + i * 32;
      const float4 f =
          *(const float4*)(X + (size_t)(m0 + r) * H + k0 + acol);
      v4bf p;
      p[0] = (bf16)f.x; p[1] = (bf16)f.y; p[2] = (bf16)f.z; p[3] = (bf16)f.w;
      *(v4bf*)(&As[r * AS_STR + acol]) = p;
    }
    if (k0 + 32 < H)  // prefetch next X K-slice (global_prefetch_b8)
      __builtin_prefetch(X + (size_t)(m0 + arow) * H + k0 + 32 + acol, 0, 0);
    // --- stage B tiles for both weights: 2 x 128x32 bf16 ---
    for (int w = 0; w < 2; ++w) {
      const bf16* Wt = w ? Wt1 : Wt0;
      for (int i = 0; i < 2; ++i) {
        const int s = tid + i * 256;     // 0..511
        const int n = s >> 2;            // 0..127
        const int kc = (s & 3) * 8;      // 0,8,16,24
        const bf16* src = Wt + (size_t)(n0 + n) * H + k0 + kc;
#if HAVE_ASYNC_LDS
        __builtin_amdgcn_global_load_async_to_lds_b128(
            gbl_cast(src), lds_cast(&Bs[w][n * AS_STR + kc]), 0, 0);
#else
        *(v8bf*)(&Bs[w][n * AS_STR + kc]) = *(const v8bf*)src;
#endif
      }
    }
#if HAVE_ASYNC_LDS
    __builtin_amdgcn_s_wait_asynccnt(0);
#endif
    __syncthreads();

    v16bf afrag[2];
    for (int fm = 0; fm < 2; ++fm) {
      const int m = waveM * 32 + fm * 16 + lane16;
      const int klo = hiHalf * 8;
      v8bf lo = *(const v8bf*)(&As[m * AS_STR + klo]);
      v8bf hi = *(const v8bf*)(&As[m * AS_STR + 16 + klo]);
      afrag[fm] = cat16(lo, hi);
    }
    for (int w = 0; w < 2; ++w)
      for (int fn = 0; fn < 4; ++fn) {
        const int n = waveN * 64 + fn * 16 + lane16;
        const int kk = hiHalf * 8;
        v8bf lo = *(const v8bf*)(&Bs[w][n * AS_STR + kk]);
        v8bf hi = *(const v8bf*)(&Bs[w][n * AS_STR + kk + 16]);
        const v16bf bfrag = cat16(lo, hi);
        acc[w][0][fn] = wmma_bf16(afrag[0], bfrag, acc[w][0][fn]);
        acc[w][1][fn] = wmma_bf16(afrag[1], bfrag, acc[w][1][fn]);
      }
    __syncthreads();
  }

  // epilogue: VGPR i -> M = i + 8*hiHalf, N = lane16
  for (int w = 0; w < 2; ++w) {
    bf16* C = w ? C1 : C0;
    for (int fm = 0; fm < 2; ++fm)
      for (int fn = 0; fn < 4; ++fn)
        for (int i = 0; i < 8; ++i) {
          const int m = m0 + waveM * 32 + fm * 16 + i + hiHalf * 8;
          const int n = n0 + waveN * 64 + fn * 16 + lane16;
          C[(size_t)m * H + n] = (bf16)acc[w][fm][fn][i];
        }
  }
}

// ---------------------------------------------------------------------------
// K2: attention scores via WMMA:
//   S[b][q][t] = scale * sum_h A[b][q][h] * K[b][t][h]   (q padded to 16)
// A: bf16 [*,16,H] (batch stride 0 for broadcast lq); K: bf16 [b][T][H] rows.
// block 256 = 8 waves, each wave a 16x16 tile; grid (T/128, B)
// ---------------------------------------------------------------------------
#define A_STR 1032

__global__ __launch_bounds__(256) void attn_scores_kernel(
    const bf16* __restrict__ Abf, size_t aBatchStride,
    const bf16* __restrict__ Kmat, float* __restrict__ Sout,
    int qCount, float scale) {
  __shared__ bf16 Al[16 * A_STR];
  const int b = blockIdx.y;
  const int t0 = blockIdx.x * 128;
  const int tid = threadIdx.x, wave = tid >> 5, lane = tid & 31;
  const int lane16 = lane & 15, hiHalf = lane >> 4;

  for (int i = 0; i < 8; ++i) {
    const int chunk = tid + i * 256;      // 0..2047
    const int q = chunk >> 7;             // 0..15
    const int hc = (chunk & 127) * 8;     // 0..1016
    *(v8bf*)(&Al[q * A_STR + hc]) =
        *(const v8bf*)(Abf + aBatchStride * (size_t)b + (size_t)q * H + hc);
  }
  __syncthreads();

  v8f acc = (v8f)(0.0f);
  const int t = t0 + wave * 16 + lane16;
  const bf16* kp = Kmat + ((size_t)b * T_SEQ + t) * H;

  for (int k0 = 0; k0 < H; k0 += 32) {
    const int klo = hiHalf * 8;
    v8bf alo = *(const v8bf*)(&Al[lane16 * A_STR + k0 + klo]);
    v8bf ahi = *(const v8bf*)(&Al[lane16 * A_STR + k0 + 16 + klo]);
    const int kk = k0 + hiHalf * 8;
    v8bf blo = *(const v8bf*)(kp + kk);
    v8bf bhi = *(const v8bf*)(kp + kk + 16);
    acc = wmma_bf16(cat16(alo, ahi), cat16(blo, bhi), acc);
  }

  for (int i = 0; i < 8; ++i) {
    const int q = i + hiHalf * 8;
    if (q < qCount)
      Sout[((size_t)b * qCount + q) * T_SEQ + t] = acc[i] * scale;
  }
}

// ---------------------------------------------------------------------------
// K3: in-place softmax over rows of length 2048 (grid = nRows, block 256)
// ---------------------------------------------------------------------------
__global__ __launch_bounds__(256) void softmax_rows_kernel(float* data) {
  float* p = data + (size_t)blockIdx.x * T_SEQ;
  __shared__ float red[256];
  const int tid = threadIdx.x;
  float v[8];
  float mx = -INFINITY;
  for (int i = 0; i < 8; ++i) {
    v[i] = p[tid + i * 256];
    mx = fmaxf(mx, v[i]);
  }
  red[tid] = mx; __syncthreads();
  for (int s = 128; s > 0; s >>= 1) {
    if (tid < s) red[tid] = fmaxf(red[tid], red[tid + s]);
    __syncthreads();
  }
  mx = red[0]; __syncthreads();
  float sum = 0.f;
  for (int i = 0; i < 8; ++i) { v[i] = __expf(v[i] - mx); sum += v[i]; }
  red[tid] = sum; __syncthreads();
  for (int s = 128; s > 0; s >>= 1) {
    if (tid < s) red[tid] += red[tid + s];
    __syncthreads();
  }
  const float inv = 1.0f / red[0];
  for (int i = 0; i < 8; ++i) p[tid + i * 256] = v[i] * inv;
}

// ---------------------------------------------------------------------------
// K4: Out[b][q][h] = sum_t W[b][q][t] * V[b][t][h]   (V bf16)
// grid (H/256, B), block 256
// ---------------------------------------------------------------------------
template <int QC>
__global__ __launch_bounds__(256) void weighted_sum_kernel(
    const float* __restrict__ W, const bf16* __restrict__ V,
    float* __restrict__ Out) {
  __shared__ float wl[QC * 64];
  const int b = blockIdx.y;
  const int h = blockIdx.x * 256 + threadIdx.x;
  float acc[QC];
  for (int q = 0; q < QC; ++q) acc[q] = 0.f;
  for (int tc = 0; tc < T_SEQ; tc += 64) {
    __syncthreads();
    for (int idx = threadIdx.x; idx < QC * 64; idx += 256)
      wl[idx] = W[((size_t)b * QC + idx / 64) * T_SEQ + tc + (idx & 63)];
    __syncthreads();
    for (int tt = 0; tt < 64; ++tt) {
      const float vv = (float)V[((size_t)b * T_SEQ + tc + tt) * H + h];
      for (int q = 0; q < QC; ++q) acc[q] += wl[q * 64 + tt] * vv;
    }
  }
  for (int q = 0; q < QC; ++q)
    Out[((size_t)b * QC + q) * H + h] = acc[q];
}

// ---------------------------------------------------------------------------
// K5: small row-vector GEMM  Out[r][n] = sum_k Ain[r*strideA+k] * W[k][n]
// grid (H/256, R), block 256
// ---------------------------------------------------------------------------
__global__ __launch_bounds__(256) void rowvec_gemm_kernel(
    const float* __restrict__ Ain, size_t strideA,
    const float* __restrict__ W, float* __restrict__ Out) {
  __shared__ float al[H];
  const int r = blockIdx.y;
  const int n = blockIdx.x * 256 + threadIdx.x;
  for (int i = threadIdx.x; i < H; i += 256) al[i] = Ain[strideA * r + i];
  __syncthreads();
  float acc = 0.f;
  for (int k = 0; k < H; ++k) acc += al[k] * W[(size_t)k * H + n];
  Out[(size_t)r * H + n] = acc;
}

// K5b: Sq[b][q][n] = sum_k Seed[b][1+q][k] * Wpq[k][n], q in 0..8
__global__ __launch_bounds__(256) void sq_gemm_kernel(
    const float* __restrict__ Seed, const float* __restrict__ Wpq,
    float* __restrict__ Sq) {
  __shared__ float sl[9 * H];
  const int b = blockIdx.y;
  const int n = blockIdx.x * 256 + threadIdx.x;
  for (int i = threadIdx.x; i < 9 * H; i += 256)
    sl[i] = Seed[(size_t)b * 10 * H + H + i];
  __syncthreads();
  float acc[9];
  for (int q = 0; q < 9; ++q) acc[q] = 0.f;
  for (int k = 0; k < H; ++k) {
    const float w = Wpq[(size_t)k * H + n];
    for (int q = 0; q < 9; ++q) acc[q] += sl[q * H + k] * w;
  }
  for (int q = 0; q < 9; ++q) Sq[((size_t)b * 9 + q) * H + n] = acc[q];
}

// ---------------------------------------------------------------------------
// K6: convert + zero-pad queries to bf16 [batches][16][H]; grid (16, batches)
// ---------------------------------------------------------------------------
__global__ void pad_bf16_kernel(const float* __restrict__ src,
                                bf16* __restrict__ dst, int qCount) {
  const int bb = blockIdx.y;
  const int q = blockIdx.x;
  for (int h = threadIdx.x; h < H; h += 256)
    dst[((size_t)bb * 16 + q) * H + h] =
        (q < qCount) ? (bf16)src[((size_t)bb * qCount + q) * H + h]
                     : (bf16)0.0f;
}

// ---------------------------------------------------------------------------
// K7: slot gates + token ids; grid (B*9)
// ---------------------------------------------------------------------------
__global__ __launch_bounds__(256) void gate_kernel(
    const float* __restrict__ Seed, const float* __restrict__ Wg,
    const float* __restrict__ bg, float* logits, float* probs,
    float* slotTok, float* matTok) {
  const int blk = blockIdx.x;
  const int b = blk / 9, q = blk % 9;
  __shared__ float red[256];
  const float* s = Seed + ((size_t)b * 10 + 1 + q) * H;
  float acc = 0.f;
  for (int h = threadIdx.x; h < H; h += 256) acc += s[h] * Wg[h];
  red[threadIdx.x] = acc; __syncthreads();
  for (int st = 128; st > 0; st >>= 1) {
    if (threadIdx.x < st) red[threadIdx.x] += red[threadIdx.x + st];
    __syncthreads();
  }
  if (threadIdx.x == 0) {
    const float l = red[0] + bg[0];
    const float p = 1.0f / (1.0f + __expf(-l));
    logits[b * 9 + q] = l;
    probs[b * 9 + q] = p;
    const float tok = (p >= 0.5f) ? (float)(2001 + q) : 2000.0f;
    slotTok[b * 9 + q] = tok;
    matTok[b * 10 + 1 + q] = tok;
  }
}

// ---------------------------------------------------------------------------
// K8: op_logits[b][c] = -||op_pre[b] - cb[c]||^2 ; grid (250, B), wave per c
// ---------------------------------------------------------------------------
__global__ __launch_bounds__(256) void cb_dist_kernel(
    const float* __restrict__ OpPre, const float* __restrict__ Cb,
    float* __restrict__ Logits) {
  __shared__ float ql[H];
  const int b = blockIdx.y;
  for (int i = threadIdx.x; i < H; i += 256) ql[i] = OpPre[(size_t)b * H + i];
  __syncthreads();
  const int wave = threadIdx.x >> 5, lane = threadIdx.x & 31;
  const int c = blockIdx.x * 8 + wave;
  const float* cp = Cb + (size_t)c * H;
  float acc = 0.f;
  for (int h0 = 0; h0 < H; h0 += 128) {
    const float4 cv = *(const float4*)(cp + h0 + lane * 4);
    const float4 qv = *(const float4*)(&ql[h0 + lane * 4]);
    const float dx = qv.x - cv.x, dy = qv.y - cv.y;
    const float dz = qv.z - cv.z, dw = qv.w - cv.w;
    acc += dx * dx + dy * dy + dz * dz + dw * dw;
  }
  for (int m = 16; m > 0; m >>= 1) acc += __shfl_xor(acc, m, 32);
  if (lane == 0) Logits[(size_t)b * CODEBOOK + c] = -acc;
}

// ---------------------------------------------------------------------------
// K9: codebook softmax + argmax + embedding gather; grid (B)
// ---------------------------------------------------------------------------
__global__ __launch_bounds__(256) void cb_softmax_kernel(
    const float* __restrict__ Logits, const float* __restrict__ Cb,
    float* Probs, float* OpIdx, float* OpEmb, float* MatTok) {
  const int b = blockIdx.x, tid = threadIdx.x;
  __shared__ float sv[256];
  __shared__ int si[256];
  __shared__ float bmax, bsum;
  __shared__ int bidx;
  const float* lp = Logits + (size_t)b * CODEBOOK;
  float mx = -INFINITY; int mi = 0;
  for (int c = tid; c < CODEBOOK; c += 256) {
    const float v = lp[c];
    if (v > mx) { mx = v; mi = c; }
  }
  sv[tid] = mx; si[tid] = mi; __syncthreads();
  for (int s = 128; s > 0; s >>= 1) {
    if (tid < s) {
      const float v2 = sv[tid + s]; const int i2 = si[tid + s];
      if (v2 > sv[tid] || (v2 == sv[tid] && i2 < si[tid])) {
        sv[tid] = v2; si[tid] = i2;
      }
    }
    __syncthreads();
  }
  if (tid == 0) { bmax = sv[0]; bidx = si[0]; }
  __syncthreads();
  float sum = 0.f;
  for (int c = tid; c < CODEBOOK; c += 256) sum += __expf(lp[c] - bmax);
  sv[tid] = sum; __syncthreads();
  for (int s = 128; s > 0; s >>= 1) {
    if (tid < s) sv[tid] += sv[tid + s];
    __syncthreads();
  }
  if (tid == 0) bsum = sv[0];
  __syncthreads();
  const float inv = 1.0f / bsum;
  for (int c = tid; c < CODEBOOK; c += 256)
    Probs[(size_t)b * CODEBOOK + c] = __expf(lp[c] - bmax) * inv;
  if (tid == 0) {
    OpIdx[b] = (float)bidx;
    MatTok[b * 10] = (float)bidx;
  }
  const int idx = bidx;
  for (int h = tid; h < H; h += 256)
    OpEmb[(size_t)b * H + h] = Cb[(size_t)idx * H + h];
}

// ---------------------------------------------------------------------------
// Host launch
// ---------------------------------------------------------------------------
extern "C" void kernel_launch(void* const* d_in, const int* in_sizes, int n_in,
                              void* d_out, int out_size, void* d_ws,
                              size_t ws_size, hipStream_t stream) {
  const float* logic_hidden  = (const float*)d_in[0];
  const float* prompt_hidden = (const float*)d_in[1];
  const float* codebook_emb  = (const float*)d_in[2];
  const float* slot_queries  = (const float*)d_in[3];
  const float* W_lq = (const float*)d_in[4];
  const float* W_lk = (const float*)d_in[5];
  const float* W_lv = (const float*)d_in[6];
  const float* W_pq = (const float*)d_in[7];
  const float* W_pk = (const float*)d_in[8];
  const float* W_pv = (const float*)d_in[9];
  const float* W_op = (const float*)d_in[10];
  const float* W_gate = (const float*)d_in[11];
  const float* b_gate = (const float*)d_in[12];

  // ---- output offsets (floats, in return order) ----
  float* out = (float*)d_out;
  size_t off = 0;
  float* o_op_pre   = out + off; off += (size_t)NB * H;
  float* o_logits   = out + off; off += (size_t)NB * CODEBOOK;
  float* o_probs    = out + off; off += (size_t)NB * CODEBOOK;
  float* o_idx      = out + off; off += NB;
  float* o_emb      = out + off; off += (size_t)NB * H;
  float* o_slot_t   = out + off; off += (size_t)NB * 9 * H;
  float* o_sul      = out + off; off += (size_t)NB * 9;
  float* o_sup      = out + off; off += (size_t)NB * 9;
  float* o_stok     = out + off; off += (size_t)NB * 9;
  float* o_mtok     = out + off; off += (size_t)NB * 10;
  float* o_lattn    = out + off; off += (size_t)NB * 10 * T_SEQ;
  float* o_pattn    = out + off; off += (size_t)NB * 9 * T_SEQ;

  // ---- workspace carving ----
  char* wsp = (char*)d_ws;
  auto carve = [&](size_t bytes) -> void* {
    void* p = (void*)wsp;
    wsp += (bytes + 255) & ~(size_t)255;
    return p;
  };
  bf16* Wt_lk = (bf16*)carve((size_t)H * H * 2);
  bf16* Wt_lv = (bf16*)carve((size_t)H * H * 2);
  bf16* Wt_pk = (bf16*)carve((size_t)H * H * 2);
  bf16* Wt_pv = (bf16*)carve((size_t)H * H * 2);
  float* lq_f32   = (float*)carve((size_t)10 * H * 4);
  bf16* lq_pad    = (bf16*)carve((size_t)16 * H * 2);
  bf16* sq_pad    = (bf16*)carve((size_t)NB * 16 * H * 2);
  float* slot_seed = (float*)carve((size_t)NB * 10 * H * 4);
  float* sq_f32    = (float*)carve((size_t)NB * 9 * H * 4);
  bf16* lk = (bf16*)carve((size_t)NB * T_SEQ * H * 2);
  bf16* lv = (bf16*)carve((size_t)NB * T_SEQ * H * 2);
  bf16* pk = (bf16*)carve((size_t)NB * T_SEQ * H * 2);
  bf16* pv = (bf16*)carve((size_t)NB * T_SEQ * H * 2);

  const float scale = 1.0f / 32.0f;  // 1/sqrt(1024)
  const int R = NB * T_SEQ;          // 131072 rows

  dim3 tb(32, 8);
  // 1) weight transpose+convert
  wt_convert_kernel<<<dim3(32, 32), tb, 0, stream>>>(W_lk, Wt_lk);
  wt_convert_kernel<<<dim3(32, 32), tb, 0, stream>>>(W_lv, Wt_lv);
  wt_convert_kernel<<<dim3(32, 32), tb, 0, stream>>>(W_pk, Wt_pk);
  wt_convert_kernel<<<dim3(32, 32), tb, 0, stream>>>(W_pv, Wt_pv);

  // 2) lq = slot_queries @ W_lq  (10 x H)
  rowvec_gemm_kernel<<<dim3(H / 256, 10), 256, 0, stream>>>(
      slot_queries, (size_t)H, W_lq, lq_f32);

  // 3) big projections: fused K/V WMMA GEMMs (one X pass each)
  gemm_kv_xf32<<<dim3(H / 128, R / 128), 256, 0, stream>>>(
      logic_hidden, Wt_lk, Wt_lv, lk, lv);
  gemm_kv_xf32<<<dim3(H / 128, R / 128), 256, 0, stream>>>(
      prompt_hidden, Wt_pk, Wt_pv, pk, pv);

  // 4) pad lq -> bf16 16xH (broadcast over batch)
  pad_bf16_kernel<<<dim3(16, 1), 256, 0, stream>>>(lq_f32, lq_pad, 10);

  // 5) logic attention scores (WMMA) -> logic_attn region, then softmax
  attn_scores_kernel<<<dim3(T_SEQ / 128, NB), 256, 0, stream>>>(
      lq_pad, (size_t)0, lk, o_lattn, 10, scale);
  softmax_rows_kernel<<<NB * 10, 256, 0, stream>>>(o_lattn);

  // 6) slot_seed = logic_attn @ lv
  weighted_sum_kernel<10><<<dim3(H / 256, NB), 256, 0, stream>>>(o_lattn, lv,
                                                                 slot_seed);

  // 7) op_pre = slot_seed[:,0,:] @ W_op
  rowvec_gemm_kernel<<<dim3(H / 256, NB), 256, 0, stream>>>(
      slot_seed, (size_t)(10 * H), W_op, o_op_pre);

  // 8) sq = slot_seed[:,1:,:] @ W_pq
  sq_gemm_kernel<<<dim3(H / 256, NB), 256, 0, stream>>>(slot_seed, W_pq,
                                                        sq_f32);
  pad_bf16_kernel<<<dim3(16, NB), 256, 0, stream>>>(sq_f32, sq_pad, 9);

  // 9) prompt attention scores (WMMA) + softmax
  attn_scores_kernel<<<dim3(T_SEQ / 128, NB), 256, 0, stream>>>(
      sq_pad, (size_t)(16 * H), pk, o_pattn, 9, scale);
  softmax_rows_kernel<<<NB * 9, 256, 0, stream>>>(o_pattn);

  // 10) slot_tensors = prompt_attn @ pv
  weighted_sum_kernel<9><<<dim3(H / 256, NB), 256, 0, stream>>>(o_pattn, pv,
                                                                o_slot_t);

  // 11) gates + slot tokens
  gate_kernel<<<NB * 9, 256, 0, stream>>>(slot_seed, W_gate, b_gate, o_sul,
                                          o_sup, o_stok, o_mtok);

  // 12) codebook distance logits, softmax, argmax, embedding gather
  cb_dist_kernel<<<dim3(CODEBOOK / 8, NB), 256, 0, stream>>>(o_op_pre,
                                                             codebook_emb,
                                                             o_logits);
  cb_softmax_kernel<<<NB, 256, 0, stream>>>(o_logits, codebook_emb, o_probs,
                                            o_idx, o_emb, o_mtok);

  (void)in_sizes; (void)n_in; (void)out_size; (void)ws_size;
}
